// CoSSM_16638703305493
// MI455X (gfx1250) — compile-verified, matrix operand order checked by reading, not run on hardware
//
#include <hip/hip_runtime.h>
#include <cmath>

// ---- problem constants ----
#define BB 8
#define LL 512
static const int DIMS[3] = {256, 512, 512};

typedef __attribute__((ext_vector_type(16))) __bf16 v16bf;
typedef __attribute__((ext_vector_type(8)))  __bf16 v8bf;
typedef __attribute__((ext_vector_type(8)))  float  v8f;
typedef __attribute__((ext_vector_type(4)))  int    v4i;

__device__ __forceinline__ v8f wmma_bf16(v16bf a, v16bf b, v8f c) {
    // D = A(16x32 bf16) * B(32x16 bf16) + C(16x16 f32)
    return __builtin_amdgcn_wmma_f32_16x16x32_bf16(
        false, a, false, b, (short)0, c, false, false);
}

// bf16 fragment load: per ISA layout lane half (lane&16) selects kb=8;
// elements 0..7 -> K=kb+i, elements 8..15 -> K=kb+16+i.
// Two 16B loads (8 bf16 each); requires 16B alignment (ld multiple of 8 elems).
__device__ __forceinline__ v16bf frag_ld(const __bf16* __restrict__ row,
                                         int k0, int lane) {
    int kb = k0 + ((lane & 16) ? 8 : 0);
    v8bf lo = *(const v8bf*)(row + kb);
    v8bf hi = *(const v8bf*)(row + kb + 16);
    return __builtin_shufflevector(lo, hi, 0, 1, 2, 3, 4, 5, 6, 7,
                                   8, 9, 10, 11, 12, 13, 14, 15);
}

// Branchless masked variant (conv sequence edges): AND raw bits with 0/-1.
__device__ __forceinline__ v16bf frag_ld_mask(const __bf16* __restrict__ row,
                                              int k0, int lane, int msk) {
    int kb = k0 + ((lane & 16) ? 8 : 0);
    v4i lo = *(const v4i*)(row + kb);
    v4i hi = *(const v4i*)(row + kb + 16);
    lo &= msk;
    hi &= msk;
    v8bf l = __builtin_bit_cast(v8bf, lo);
    v8bf h = __builtin_bit_cast(v8bf, hi);
    return __builtin_shufflevector(l, h, 0, 1, 2, 3, 4, 5, 6, 7,
                                   8, 9, 10, 11, 12, 13, 14, 15);
}

// ------- f32 -> bf16 elementwise convert -------
__global__ __launch_bounds__(256) void k_cvt_bf16(const float* __restrict__ src,
                                                  __bf16* __restrict__ dst, size_t n) {
    size_t i = (size_t)blockIdx.x * 256 + threadIdx.x;
    if (i < n) dst[i] = (__bf16)src[i];
}

// ------- f32 (rows, sc) -> bf16 (rows, dc) convert with zero right-pad -------
__global__ __launch_bounds__(256) void k_cvt_pad_rows(const float* __restrict__ src,
                                                      __bf16* __restrict__ dst,
                                                      int sc, int dc, size_t n) {
    size_t i = (size_t)blockIdx.x * 256 + threadIdx.x;
    if (i >= n) return;
    int c = (int)(i % dc);
    size_t r = i / dc;
    dst[i] = (c < sc) ? (__bf16)src[r * sc + c] : (__bf16)0.0f;
}

// ------- repack conv weights (d, prev, 3) -> 3 contiguous (d, prev) bf16 taps -------
__global__ __launch_bounds__(256) void k_repack_conv(const float* __restrict__ cw,
                                                     __bf16* __restrict__ wt,
                                                     int prev, int d) {
    size_t idx = (size_t)blockIdx.x * 256 + threadIdx.x;
    size_t n = (size_t)d * prev;
    if (idx >= 3 * n) return;
    int t = (int)(idx / n);
    size_t rem = idx - (size_t)t * n;   // m*prev + c
    wt[idx] = (__bf16)cw[rem * 3 + t];
}

// Conv inner K-loop, compile-time specialized on edge masking and skip GEMM.
template <bool MASKED, bool SKIP>
__device__ __forceinline__ void conv_loop(
        const __bf16* __restrict__ ar0, const __bf16* __restrict__ ar1,
        const __bf16* __restrict__ ar2, int mk0, int mk2,
        const __bf16* __restrict__ brow0, const __bf16* __restrict__ srow,
        size_t tapsz, int prev, int lane, v8f& acc, v8f& accs) {
    for (int k0 = 0; k0 < prev; k0 += 32) {
        v16bf a1 = frag_ld(ar1, k0, lane);                 // center tap, always valid
        v16bf a0 = MASKED ? frag_ld_mask(ar0, k0, lane, mk0) : frag_ld(ar0, k0, lane);
        v16bf b0 = frag_ld(brow0, k0, lane);
        acc = wmma_bf16(a0, b0, acc);
        v16bf b1 = frag_ld(brow0 + tapsz, k0, lane);
        acc = wmma_bf16(a1, b1, acc);
        v16bf a2 = MASKED ? frag_ld_mask(ar2, k0, lane, mk2) : frag_ld(ar2, k0, lane);
        v16bf b2 = frag_ld(brow0 + 2 * tapsz, k0, lane);
        acc = wmma_bf16(a2, b2, acc);
        if (SKIP) {
            v16bf bs = frag_ld(srow, k0, lane);
            accs = wmma_bf16(a1, bs, accs);                // reuse center-tap A
        }
    }
}

// ------- fused Conv1d(k=3,pad=1) + BN + ReLU + skip, WMMA -------
// xf: (B,L,prev) f32 (identity skip), xb: same in bf16 (WMMA A side).
// wt: repacked bf16 taps [t][m][c]. skip_w: bf16 (d,prev) when SKIP.
// out: f32 (B,L,d) residual copy; outb: bf16 copy (in_proj A side).
// GEMM orientation: M = tokens, N = out channels, K = prev.
template <bool SKIP>
__global__ __launch_bounds__(256) void k_conv_bn_relu_skip(
        const float* __restrict__ xf, const __bf16* __restrict__ xb,
        const __bf16* __restrict__ wt, const __bf16* __restrict__ skip_w,
        const float* __restrict__ bg, const float* __restrict__ bb,
        const float* __restrict__ bm, const float* __restrict__ bv,
        float* __restrict__ out, __bf16* __restrict__ outb, int prev, int d) {
    int lane = threadIdx.x & 31;
    int wave = threadIdx.x >> 5;
    int nt = blockIdx.x;                 // d/16 channel tile
    int mt = blockIdx.y * 8 + wave;      // token tile
    int rn = lane & 15;

    int tok = mt * 16 + rn;              // this lane's A row (token)
    int b = tok >> 9;                    // / LL
    int l = tok & (LL - 1);
    int l0t = (mt * 16) & (LL - 1);      // tile's first position (wave-uniform)

    const __bf16* ar[3];
    int mk[3];
#pragma unroll
    for (int t = 0; t < 3; ++t) {
        int ls = l + t - 1;
        mk[t] = (ls >= 0 && ls < LL) ? -1 : 0;
        ls = min(max(ls, 0), LL - 1);
        ar[t] = xb + ((size_t)(b << 9) + ls) * prev;
    }
    size_t tapsz = (size_t)d * prev;
    const __bf16* brow0 = wt + (size_t)(nt * 16 + rn) * prev;
    const __bf16* srow  = SKIP ? skip_w + (size_t)(nt * 16 + rn) * prev : nullptr;

    v8f acc  = {};
    v8f accs = {};
    // 30 of 32 L-tiles never touch a sequence boundary: take the unmasked path.
    if (l0t >= 16 && l0t <= LL - 32)
        conv_loop<false, SKIP>(ar[0], ar[1], ar[2], mk[0], mk[2],
                               brow0, srow, tapsz, prev, lane, acc, accs);
    else
        conv_loop<true, SKIP>(ar[0], ar[1], ar[2], mk[0], mk[2],
                              brow0, srow, tapsz, prev, lane, acc, accs);

    // epilogue: col fixed per lane; 8 token rows per lane
    int hi = (lane & 16) ? 8 : 0;
    int col = nt * 16 + rn;
    float inv = rsqrtf(bv[col] + 1e-5f);
    float sc = inv * bg[col];
    float mean = bm[col], beta = bb[col];
#pragma unroll
    for (int v = 0; v < 8; ++v) {
        int tr = mt * 16 + hi + v;
        float val = fmaxf((acc[v] - mean) * sc + beta, 0.0f);
        float sk = SKIP ? accs[v] : xf[(size_t)tr * prev + col];
        float o = val + sk;
        out[(size_t)tr * d + col]  = o;
        outb[(size_t)tr * d + col] = (__bf16)o;
    }
}

// ------- generic WMMA GEMM (16x16/wave): Out[r,c] = sum_k X[r,k] * W[c,k] -------
// K multiple of 32.  ACT: 0 none, 1 softplus (bias f32, per column, nullable).
template <int ACT>
__global__ __launch_bounds__(256) void k_gemm(
        const __bf16* __restrict__ X, int lda,
        const __bf16* __restrict__ W, int ldw,
        float* __restrict__ Out, int ldo,
        int K, const float* __restrict__ bias) {
    int lane = threadIdx.x & 31;
    int wave = threadIdx.x >> 5;
    int nt = blockIdx.x;
    int mt = blockIdx.y * 8 + wave;
    int rn = lane & 15;

    const __bf16* xrow = X + (size_t)(mt * 16 + rn) * lda;
    const __bf16* wrow = W + (size_t)(nt * 16 + rn) * ldw;
    v8f acc = {};
    for (int k0 = 0; k0 < K; k0 += 32) {
        __builtin_prefetch(xrow + k0 + 32, 0, 0);
        v16bf a = frag_ld(xrow, k0, lane);
        v16bf b = frag_ld(wrow, k0, lane);
        acc = wmma_bf16(a, b, acc);
    }
    int hi = (lane & 16) ? 8 : 0;
    int col = nt * 16 + rn;
    float bvv = bias ? bias[col] : 0.0f;
#pragma unroll
    for (int v = 0; v < 8; ++v) {
        int row = mt * 16 + hi + v;
        float val = acc[v] + bvv;
        if (ACT == 1) val = (val > 20.0f) ? val : log1pf(__expf(val));
        Out[(size_t)row * ldo + col] = val;
    }
}

// ------- 2x2 register-blocked WMMA GEMM (32x32/wave), no bias/act -------
__global__ __launch_bounds__(256) void k_gemm_2x2(
        const __bf16* __restrict__ X, int lda,
        const __bf16* __restrict__ W, int ldw,
        float* __restrict__ Out, int ldo, int K) {
    int lane = threadIdx.x & 31;
    int wave = threadIdx.x >> 5;
    int nt = blockIdx.x;                 // N/32
    int mt = blockIdx.y * 8 + wave;      // M/32
    int rn = lane & 15;

    const __bf16* xr0 = X + (size_t)(mt * 32 + rn) * lda;
    const __bf16* xr1 = xr0 + (size_t)16 * lda;
    const __bf16* wr0 = W + (size_t)(nt * 32 + rn) * ldw;
    const __bf16* wr1 = wr0 + (size_t)16 * ldw;

    v8f a00 = {}, a01 = {}, a10 = {}, a11 = {};
    for (int k0 = 0; k0 < K; k0 += 32) {
        __builtin_prefetch(xr0 + k0 + 32, 0, 0);
        __builtin_prefetch(xr1 + k0 + 32, 0, 0);
        v16bf fa0 = frag_ld(xr0, k0, lane);
        v16bf fa1 = frag_ld(xr1, k0, lane);
        v16bf fb0 = frag_ld(wr0, k0, lane);
        v16bf fb1 = frag_ld(wr1, k0, lane);
        a00 = wmma_bf16(fa0, fb0, a00);
        a01 = wmma_bf16(fa0, fb1, a01);
        a10 = wmma_bf16(fa1, fb0, a10);
        a11 = wmma_bf16(fa1, fb1, a11);
    }
    int hi = (lane & 16) ? 8 : 0;
    int col0 = nt * 32 + rn;
    int row0 = mt * 32 + hi;
#pragma unroll
    for (int v = 0; v < 8; ++v) {
        Out[(size_t)(row0 + v) * ldo + col0]           = a00[v];
        Out[(size_t)(row0 + v) * ldo + col0 + 16]      = a01[v];
        Out[(size_t)(row0 + 16 + v) * ldo + col0]      = a10[v];
        Out[(size_t)(row0 + 16 + v) * ldo + col0 + 16] = a11[v];
    }
}

// ------- pad dt columns of dbc (BL, R+32) -> bf16 dtpad (BL, 32) -------
__global__ __launch_bounds__(256) void k_pad_dt(const float* __restrict__ dbc,
                                                __bf16* __restrict__ dtpad, int R) {
    size_t idx = (size_t)blockIdx.x * 256 + threadIdx.x;   // over BL*32
    int col = (int)(idx & 31);
    size_t row = idx >> 5;
    dtpad[idx] = (col < R) ? (__bf16)dbc[row * (size_t)(R + 32) + col] : (__bf16)0.0f;
}

// ------- depthwise causal conv (k=4) + SiLU; dual f32/bf16 output -------
__global__ __launch_bounds__(256) void k_dwconv_silu(
        const float* __restrict__ xz, const float* __restrict__ w,
        const float* __restrict__ bias, float* __restrict__ xc,
        __bf16* __restrict__ xcb, int din) {
    size_t idx = (size_t)blockIdx.x * 256 + threadIdx.x;
    if (idx >= (size_t)BB * LL * din) return;
    int e = (int)(idx % din);
    size_t bl = idx / din;
    int l = (int)(bl % LL);
    float acc = bias[e];
#pragma unroll
    for (int j = 0; j < 4; ++j) {
        int ls = l + j - 3;
        if (ls >= 0)
            acc += w[e * 4 + j] * xz[(bl - l + ls) * (size_t)(2 * din) + e];
    }
    float s = 1.0f / (1.0f + __expf(-acc));
    float o = acc * s;
    xc[idx]  = o;
    xcb[idx] = (__bf16)o;
}

// ------- selective scan (sequential over L); writes bf16 y (out_proj A side) -------
__global__ __launch_bounds__(256) void k_scan(
        const float* __restrict__ xz, const float* __restrict__ xconv,
        const float* __restrict__ dbc, const float* __restrict__ dt,
        const float* __restrict__ A_log, const float* __restrict__ Dv,
        __bf16* __restrict__ y, int din, int R) {
    int idx = blockIdx.x * 256 + threadIdx.x;   // b*din + e
    if (idx >= BB * din) return;
    int b = idx / din;
    int e = idx % din;
    float A[16], h[16];
#pragma unroll
    for (int n = 0; n < 16; ++n) { A[n] = -__expf(A_log[(size_t)e * 16 + n]); h[n] = 0.0f; }
    float dskip = Dv[e];
    int ldbc = R + 32;
    for (int l = 0; l < LL; ++l) {
        size_t t = (size_t)b * LL + l;
        float dtv = dt[t * din + e];
        float xv  = xconv[t * din + e];
        const float* bc = dbc + t * ldbc + R;      // B at [0..15], C at [16..31]
        float acc = 0.0f;
#pragma unroll
        for (int n = 0; n < 16; ++n) {
            float dA = __expf(dtv * A[n]);
            h[n] = dA * h[n] + dtv * bc[n] * xv;
            acc += h[n] * bc[16 + n];
        }
        acc += xv * dskip;
        float zv = xz[t * (size_t)(2 * din) + din + e];
        float sg = zv / (1.0f + __expf(-zv));
        y[t * din + e] = (__bf16)(acc * sg);
    }
}

// ------- LayerNorm(mamba_out)*g + b + residual; dual f32/bf16 output -------
__global__ __launch_bounds__(256) void k_ln_residual(
        const float* __restrict__ resid, const float* __restrict__ xm,
        const float* __restrict__ g, const float* __restrict__ be,
        float* __restrict__ out, __bf16* __restrict__ outb, int d) {
    int t = blockIdx.x;            // token index b*L + l
    int tid = threadIdx.x;
    const float* row = xm + (size_t)t * d;
    float sum = 0.0f, sq = 0.0f;
    for (int c = tid; c < d; c += 256) { float v = row[c]; sum += v; sq += v * v; }
#pragma unroll
    for (int off = 16; off > 0; off >>= 1) {
        sum += __shfl_down(sum, off);
        sq  += __shfl_down(sq,  off);
    }
    __shared__ float s1[8], s2[8];
    int lane = tid & 31, wave = tid >> 5;
    if (lane == 0) { s1[wave] = sum; s2[wave] = sq; }
    __syncthreads();
    if (tid == 0) {
        float a = 0.0f, b2 = 0.0f;
        for (int i = 0; i < 8; ++i) { a += s1[i]; b2 += s2[i]; }
        s1[0] = a; s2[0] = b2;
    }
    __syncthreads();
    float mu  = s1[0] / d;
    float var = s2[0] / d - mu * mu;
    float rs  = rsqrtf(var + 1e-6f);
    for (int c = tid; c < d; c += 256) {
        float o = resid[(size_t)t * d + c] + (row[c] - mu) * rs * g[c] + be[c];
        out[(size_t)t * d + c]  = o;
        outb[(size_t)t * d + c] = (__bf16)o;
    }
}

extern "C" void kernel_launch(void* const* d_in, const int* in_sizes, int n_in,
                              void* d_out, int out_size, void* d_ws, size_t ws_size,
                              hipStream_t stream) {
    (void)in_sizes; (void)n_in; (void)out_size; (void)ws_size;
    int idx = 0;
    auto nxt = [&]() { return (const float*)d_in[idx++]; };
    const float* x_in = nxt();

    char* wp = (char*)d_ws;
    auto carve = [&](size_t bytes) {
        void* p = wp;
        wp += (bytes + 255) & ~(size_t)255;
        return p;
    };
    const size_t BL = (size_t)BB * LL;
    // f32 buffers
    float* act   = (float*)carve(BL * 512 * 4);
    float* cnn   = (float*)carve(BL * 512 * 4);
    float* xz    = (float*)carve(BL * 2048 * 4);
    float* xconv = (float*)carve(BL * 1024 * 4);
    float* dbc   = (float*)carve(BL * 64 * 4);
    float* dtb   = (float*)carve(BL * 1024 * 4);
    float* mam   = (float*)carve(BL * 512 * 4);
    // bf16 buffers
    __bf16* xinbf   = (__bf16*)carve(BL * 256 * 2);
    __bf16* actbf   = (__bf16*)carve(BL * 512 * 2);
    __bf16* cnnbf   = (__bf16*)carve(BL * 512 * 2);
    __bf16* xconvbf = (__bf16*)carve(BL * 1024 * 2);
    __bf16* dtpadbf = (__bf16*)carve(BL * 32 * 2);
    __bf16* ybbf    = (__bf16*)carve(BL * 1024 * 2);
    __bf16* wtap    = (__bf16*)carve((size_t)3 * 512 * 512 * 2);
    __bf16* skipbf  = (__bf16*)carve((size_t)512 * 512 * 2);
    __bf16* inpbf   = (__bf16*)carve((size_t)2048 * 512 * 2);
    __bf16* xpbf    = (__bf16*)carve((size_t)64 * 1024 * 2);
    __bf16* dtwbf   = (__bf16*)carve((size_t)1024 * 32 * 2);
    __bf16* outpbf  = (__bf16*)carve((size_t)512 * 1024 * 2);

    auto cvt = [&](const float* s, __bf16* dst, size_t n) {
        k_cvt_bf16<<<(unsigned)((n + 255) / 256), 256, 0, stream>>>(s, dst, n);
    };

    // layer-0 input bf16 copy
    cvt(x_in, xinbf, BL * 256);

    const float*  curf = x_in;
    const __bf16* curb = xinbf;
    int prev = 256;
    for (int li = 0; li < 3; ++li) {
        int d = DIMS[li], din = 2 * d, R = (d + 15) / 16;
        const float* conv_w = nxt();
        const float* bn_g = nxt(); const float* bn_b = nxt();
        const float* bn_m = nxt(); const float* bn_v = nxt();
        const float* skip_w = (prev != d) ? nxt() : nullptr;
        const float* in_proj = nxt();
        const float* dw_w = nxt(); const float* dw_b = nxt();
        const float* x_proj = nxt();
        const float* dt_w = nxt(); const float* dt_b = nxt();
        const float* A_log = nxt(); const float* Dv = nxt();
        const float* out_proj = nxt();
        const float* ln_g = nxt(); const float* ln_b = nxt();

        // weight prep (bf16)
        k_repack_conv<<<(unsigned)(((size_t)3 * d * prev + 255) / 256), 256, 0, stream>>>(
            conv_w, wtap, prev, d);
        if (skip_w) cvt(skip_w, skipbf, (size_t)d * prev);
        cvt(in_proj, inpbf, (size_t)2 * din * d);
        cvt(x_proj, xpbf, (size_t)(R + 32) * din);
        {   // dt_proj weight: (din, R) -> (din, 32) zero-padded
            size_t n = (size_t)din * 32;
            k_cvt_pad_rows<<<(unsigned)((n + 255) / 256), 256, 0, stream>>>(
                dt_w, dtwbf, R, 32, n);
        }
        cvt(out_proj, outpbf, (size_t)d * din);

        // conv + bn + relu + skip -> cnn f32 + cnnbf
        dim3 gc(d / 16, (unsigned)(BL / 128));
        if (skip_w)
            k_conv_bn_relu_skip<true><<<gc, 256, 0, stream>>>(
                curf, curb, wtap, skipbf, bn_g, bn_b, bn_m, bn_v, cnn, cnnbf, prev, d);
        else
            k_conv_bn_relu_skip<false><<<gc, 256, 0, stream>>>(
                curf, curb, wtap, nullptr, bn_g, bn_b, bn_m, bn_v, cnn, cnnbf, prev, d);
        // in_proj -> xz f32 [blocked]
        k_gemm_2x2<<<dim3(2 * din / 32, (unsigned)(BL / 256)), 256, 0, stream>>>(
            cnnbf, d, inpbf, d, xz, 2 * din, d);
        // depthwise conv + silu -> xconv f32 + xconvbf
        k_dwconv_silu<<<(unsigned)((BL * din + 255) / 256), 256, 0, stream>>>(
            xz, dw_w, dw_b, xconv, xconvbf, din);
        // x_proj -> dbc f32 [generic, small N]
        k_gemm<0><<<dim3((R + 32) / 16, (unsigned)(BL / 128)), 256, 0, stream>>>(
            xconvbf, din, xpbf, din, dbc, R + 32, din, nullptr);
        // dt slice pad (bf16) + dt_proj + bias + softplus -> dtb f32
        k_pad_dt<<<(unsigned)(BL * 32 / 256), 256, 0, stream>>>(dbc, dtpadbf, R);
        k_gemm<1><<<dim3(din / 16, (unsigned)(BL / 128)), 256, 0, stream>>>(
            dtpadbf, 32, dtwbf, 32, dtb, din, 32, dt_b);
        // selective scan -> ybbf (bf16)
        k_scan<<<(BB * din + 255) / 256, 256, 0, stream>>>(
            xz, xconv, dbc, dtb, A_log, Dv, ybbf, din, R);
        // out_proj -> mam f32 [blocked]
        k_gemm_2x2<<<dim3(d / 32, (unsigned)(BL / 256)), 256, 0, stream>>>(
            ybbf, din, outpbf, din, mam, d, din);
        // residual + layernorm -> act f32 + actbf (final layer: d_out + scratch bf)
        float* dst = (li == 2) ? (float*)d_out : act;
        k_ln_residual<<<(unsigned)BL, 256, 0, stream>>>(cnn, mam, ln_g, ln_b,
                                                        dst, actbf, d);
        curf = act;
        curb = actbf;
        prev = d;
    }
}
